// Sequence_9440338116704
// MI455X (gfx1250) — matrix operand order repaired and checked by reading, not compiled, not run on hardware
//
#include <hip/hip_runtime.h>
#include <hip/hip_bf16.h>

// ---------------------------------------------------------------------------
// Sequential LSTM (T=2048, B=256, H=128, IN_DIM=1) for MI455X (gfx1250).
//
// Design:
//  * 16 workgroups x 256 threads (8 wave32s). Each workgroup owns 16 batch
//    rows (rows are fully independent -> no inter-WG sync).
//  * Per step: gates(16x512) = h(16x128) @ W_hh^T via v_wmma_f32_16x16x32_bf16.
//    Wave w owns hidden columns [16w,16w+16) of ALL four gates (i,f,g,o), so
//    the per-(row,col) LSTM cell state lives in 8 regs/lane with no exchange.
//  * W_hh is converted once to bf16 (d_ws) and its 16 WMMA B-fragments are
//    loaded into registers BEFORE the T loop -> steady state touches LDS only
//    for the 4KB h tile (bf16) and the 16-entry x vector.
//  * bias + x*W_ih folded into the WMMA C initializer. y = h . W_lin via
//    shfl_xor reduction + LDS combine; y feeds back as next x.
// ---------------------------------------------------------------------------

typedef __bf16 bf16_t;
typedef __attribute__((ext_vector_type(16))) __bf16 v16bf;
typedef __attribute__((ext_vector_type(8)))  __bf16 v8bf;
typedef __attribute__((ext_vector_type(8)))  float  v8f;

#define SEQ_T   2048
#define BATCH   256
#define HID     128
#define GATES   512     // 4*HID
#define WG_ROWS 16      // batch rows per workgroup
#define NWG     (BATCH / WG_ROWS)

union AFrag { v16bf v; v8bf h[2]; };

__device__ __forceinline__ float fast_sigmoid(float x) {
    float e = __expf(-x);
    return __builtin_amdgcn_rcpf(1.0f + e);
}
__device__ __forceinline__ float fast_tanh(float x) {
    float e = __expf(x + x);                       // e^(2x)
    return (e - 1.0f) * __builtin_amdgcn_rcpf(e + 1.0f);
}

// --- prolog: f32 -> bf16 copy of W_hh into workspace ------------------------
__global__ void cvt_whh_bf16(const float* __restrict__ w,
                             bf16_t* __restrict__ o, int n) {
    int i = blockIdx.x * blockDim.x + threadIdx.x;
    if (i < n) o[i] = (bf16_t)w[i];
}

// --- main recurrent kernel --------------------------------------------------
__global__ __launch_bounds__(256)
void lstm_seq_kernel(const float* __restrict__ h0,      // (4096,128)
                     const int*   __restrict__ label,   // (256,)
                     const float* __restrict__ W_ih,    // (512,1)
                     const float* __restrict__ b_ih,    // (512,)
                     const float* __restrict__ b_hh,    // (512,)
                     const float* __restrict__ W_lin,   // (1,128)
                     const float* __restrict__ b_lin,   // (1,)
                     const bf16_t* __restrict__ Whh_bf, // (512,128) bf16
                     float* __restrict__ out)           // (2048,256)
{
    __shared__ __align__(32) bf16_t h_sh[WG_ROWS * HID];   // 4KB current h (bf16)
    __shared__ __align__(32) float  x_sh[WG_ROWS];         // fed-back scalar input
    __shared__ __align__(16) float  y_part[8][WG_ROWS];    // per-wave y partials

    const int tid  = threadIdx.x;
    const int wave = tid >> 5;
    const int lane = tid & 31;
    const int hi   = lane >> 4;       // half-wave group (0/1)
    const int ln   = lane & 15;       // lane-in-half
    const int row0 = blockIdx.x * WG_ROWS;

    // ---- init: gather h0 rows via label, x = 0 ----
    for (int e = tid; e < WG_ROWS * HID; e += 256) {
        int m = e >> 7, k = e & (HID - 1);
        int lbl = label[row0 + m];
        h_sh[e] = (bf16_t)h0[lbl * HID + k];
    }
    if (tid < WG_ROWS) x_sh[tid] = 0.0f;

    // ---- per-lane constants ----
    const int j = wave * 16 + ln;                 // hidden column owned by lane
    const float wlin_j  = W_lin[j];
    const float blin_s  = b_lin[0];
    float bias[4], wih[4];
#pragma unroll
    for (int g = 0; g < 4; ++g) {
        int col = g * HID + j;                    // torch gate order i,f,g,o
        bias[g] = b_ih[col] + b_hh[col];
        wih[g]  = W_ih[col];
    }

    // ---- preload W_hh^T WMMA B-fragments into registers (constant over T) --
    // B[k][n] = W_hh[n][k]; lane L holds col n=L%16, K = 16*(L/16)+q (q=0..15)
    // -> 32 contiguous bf16 from row n of W_hh.
    v16bf Bfr[4][4];
#pragma unroll
    for (int g = 0; g < 4; ++g) {
#pragma unroll
        for (int kt = 0; kt < 4; ++kt) {
            int col   = g * HID + j;
            int kbase = kt * 32 + hi * 16;
            Bfr[g][kt] = *(const v16bf*)(Whh_bf + (size_t)col * HID + kbase);
        }
    }

    float c_state[8];
#pragma unroll
    for (int r = 0; r < 8; ++r) c_state[r] = 0.0f;

    __syncthreads();

    // ======================= sequential time loop ==========================
    for (int t = 0; t < SEQ_T; ++t) {
        // x values for the 8 rows this lane accumulates (rows r + 8*hi)
        const float4* xp = (const float4*)(x_sh + hi * 8);
        float4 x0 = xp[0], x1 = xp[1];
        float xv[8] = {x0.x, x0.y, x0.z, x0.w, x1.x, x1.y, x1.z, x1.w};

        // C init: bias + x * W_ih  (rank-1 input term folded into accumulator)
        v8f acc[4];
#pragma unroll
        for (int g = 0; g < 4; ++g)
#pragma unroll
            for (int r = 0; r < 8; ++r)
                acc[g][r] = fmaf(xv[r], wih[g], bias[g]);

        // gates += h @ W_hh^T : 4 K-tiles x 4 gate tiles of WMMA
#pragma unroll
        for (int kt = 0; kt < 4; ++kt) {
            AFrag a;
            const int abase = ln * HID + kt * 32 + hi * 16;
            a.h[0] = *(const v8bf*)(h_sh + abase);        // K = base .. +7
            a.h[1] = *(const v8bf*)(h_sh + abase + 8);    // K = base+8 .. +15
#pragma unroll
            for (int g = 0; g < 4; ++g)
                acc[g] = __builtin_amdgcn_wmma_f32_16x16x32_bf16(
                    false, a.v, false, Bfr[g][kt], (short)0, acc[g],
                    false, false);
        }

        // ---- LSTM cell (per-lane: rows r+8*hi, hidden col j) ----
        float hnew[8];
#pragma unroll
        for (int r = 0; r < 8; ++r) {
            float iv = fast_sigmoid(acc[0][r]);
            float fv = fast_sigmoid(acc[1][r]);
            float gv = fast_tanh   (acc[2][r]);
            float ov = fast_sigmoid(acc[3][r]);
            float cc = fmaf(fv, c_state[r], iv * gv);
            c_state[r] = cc;
            hnew[r] = ov * fast_tanh(cc);
        }

        __syncthreads();   // all reads of h_sh / x_sh for step t complete

        // write new h (bf16) + per-wave y partials
#pragma unroll
        for (int r = 0; r < 8; ++r) {
            int m = r + 8 * hi;
            h_sh[m * HID + j] = (bf16_t)hnew[r];
            float p = hnew[r] * wlin_j;
            p += __shfl_xor(p, 1);
            p += __shfl_xor(p, 2);
            p += __shfl_xor(p, 4);
            p += __shfl_xor(p, 8);
            if (ln == 0) y_part[wave][m] = p;   // lanes 0 and 16 publish
        }

        __syncthreads();   // h_sh / y_part visible

        if (tid < WG_ROWS) {
            float y = blin_s;
#pragma unroll
            for (int w = 0; w < 8; ++w) y += y_part[w][tid];
            x_sh[tid] = y;                                  // feedback input
            out[(size_t)t * BATCH + row0 + tid] = y;        // ys[t, b, 0]
        }

        __syncthreads();   // x_sh ready for step t+1
    }
}

// ---------------------------------------------------------------------------
extern "C" void kernel_launch(void* const* d_in, const int* in_sizes, int n_in,
                              void* d_out, int out_size, void* d_ws, size_t ws_size,
                              hipStream_t stream) {
    // setup_inputs order:
    // 0 input(unused) 1 label 2 h0 3 W_ih 4 W_hh 5 b_ih 6 b_hh 7 W_lin 8 b_lin
    const int*   label = (const int*)  d_in[1];
    const float* h0    = (const float*)d_in[2];
    const float* W_ih  = (const float*)d_in[3];
    const float* W_hh  = (const float*)d_in[4];
    const float* b_ih  = (const float*)d_in[5];
    const float* b_hh  = (const float*)d_in[6];
    const float* W_lin = (const float*)d_in[7];
    const float* b_lin = (const float*)d_in[8];

    bf16_t* whh_bf = (bf16_t*)d_ws;               // 512*128*2 = 128KB scratch
    const int nW = GATES * HID;
    cvt_whh_bf16<<<(nW + 255) / 256, 256, 0, stream>>>(W_hh, whh_bf, nW);

    lstm_seq_kernel<<<NWG, 256, 0, stream>>>(h0, label, W_ih, b_ih, b_hh,
                                             W_lin, b_lin, whh_bf,
                                             (float*)d_out);
}
